// Memory_81260781240792
// MI455X (gfx1250) — matrix operand order repaired and checked by reading, not compiled, not run on hardware
//
#include <hip/hip_runtime.h>
#include <hip/hip_bf16.h>
#include <math.h>

typedef __bf16 bf16_t;
typedef __bf16 v16bf __attribute__((ext_vector_type(16)));
typedef float  v8f   __attribute__((ext_vector_type(8)));
typedef unsigned int v4u __attribute__((ext_vector_type(4)));
typedef int          v8i __attribute__((ext_vector_type(8)));
typedef int          v4i __attribute__((ext_vector_type(4)));

#define B_  16
#define C_  64
#define T_  32
#define D_  512
#define M_  512
#define NR_ 32768   // B_*C_*T_

union FragBF { uint4 u[2]; v16bf v; };

__device__ __forceinline__ v8f v8f_zero() {
    v8f z = {0.f,0.f,0.f,0.f,0.f,0.f,0.f,0.f};
    return z;
}

// Flat addresses of LDS objects carry the LDS byte offset in addr[31:0].
__device__ __forceinline__ unsigned lds_offset_of(const void* p) {
    return (unsigned)(unsigned long long)(uintptr_t)p;
}

// ---------------------------------------------------------------------------
// TDM: DMA one 128-row x 32-col bf16 tile of a row-major [rows x 512] matrix
// into LDS (contiguous 128x32). Tracked by TENSORcnt.
//   D# group0: count=1 | lds_addr | global_addr(57b) | type=2
//   D# group1: data_size=1(2B), tensor_dim0=512, tensor_dim1=32768,
//              tile_dim0=32, tile_dim1=128, tensor_dim0_stride=512
// ---------------------------------------------------------------------------
__device__ __forceinline__ void tdm_load_tile_128x32(const bf16_t* gptr, unsigned lds_off) {
    unsigned long long ga = (unsigned long long)(uintptr_t)gptr;
    v4u g0;
    g0[0] = 1u;                                          // count=1, user mode
    g0[1] = lds_off;                                     // lds_addr (bytes)
    g0[2] = (unsigned)(ga & 0xffffffffu);                // global_addr[31:0]
    g0[3] = (unsigned)((ga >> 32) & 0x01ffffffu)         // global_addr[56:32]
          | (2u << 30);                                  // type=2 ("image")
    v8i g1;
    g1[0] = (int)0x00010000;       // workgroup_mask=0, data_size=1 (2 bytes)
    g1[1] = (int)(512u << 16);     // tensor_dim0[15:0]=512 @ bits 63:48
    g1[2] = (int)(0x8000u << 16);  // tensor_dim0 hi=0; tensor_dim1[15:0]=32768 @ bits 95:80
    g1[3] = (int)(32u << 16);      // tensor_dim1 hi=0; tile_dim0=32 @ bits 127:112
    g1[4] = (int)128;              // tile_dim1=128, tile_dim2=0
    g1[5] = (int)512;              // tensor_dim0_stride=512 (low 32 of 48)
    g1[6] = 0;                     // stride hi, tensor_dim1_stride lo
    g1[7] = 0;
    v4i z4 = {0,0,0,0};
#if __has_include(<hip/amd_detail/amd_gfx1250_TDM.h>)
    v8i z8 = {0,0,0,0,0,0,0,0};
    __builtin_amdgcn_tensor_load_to_lds(g0, g1, z4, z4, z8, 0);   // 6-arg toolchain
#else
    __builtin_amdgcn_tensor_load_to_lds(g0, g1, z4, z4, 0);       // 5-arg toolchain
#endif
}

// ---------------------------------------------------------------------------
// 0) zero scratch (query_update + 2 loss accumulators, contiguous)
// ---------------------------------------------------------------------------
__global__ void k_zero(float* __restrict__ p, int n) {
    int i = blockIdx.x * blockDim.x + threadIdx.x;
    for (; i < n; i += gridDim.x * blockDim.x) p[i] = 0.f;
}

// ---------------------------------------------------------------------------
// 1) inverse norms along C: inv_norm[b,t,d] = 1/max(||query[b,:,t,d]||, 1e-12)
// ---------------------------------------------------------------------------
__global__ void k_inv_norm(const float* __restrict__ q, float* __restrict__ invn) {
    int idx = blockIdx.x * blockDim.x + threadIdx.x;      // over B_*T_*D_ = 262144
    if (idx >= B_ * T_ * D_) return;
    int d = idx % D_;
    int t = (idx / D_) % T_;
    int b = idx / (D_ * T_);
    const float* p = q + ((size_t)b * C_ * T_ + t) * D_ + d;   // c = 0
    float s = 0.f;
    #pragma unroll 4
    for (int c = 0; c < C_; ++c) {
        float v = p[(size_t)c * T_ * D_];
        s += v * v;
    }
    invn[idx] = 1.0f / fmaxf(sqrtf(s), 1e-12f);
}

// ---------------------------------------------------------------------------
// 2) qr_bf16[i,d] = bf16(query[i,d] * inv_norm[b,t,d])
// ---------------------------------------------------------------------------
__global__ void k_qr_bf16(const float* __restrict__ q, const float* __restrict__ invn,
                          bf16_t* __restrict__ qr) {
    int idx = blockIdx.x * blockDim.x + threadIdx.x;      // over NR_*D_ = 16777216
    if (idx >= NR_ * D_) return;
    int d = idx % D_;
    int i = idx / D_;
    int t = i % T_;
    int b = i / (C_ * T_);
    float v = q[idx] * invn[((size_t)(b * T_ + t)) * D_ + d];
    qr[idx] = (bf16_t)v;
}

// ---------------------------------------------------------------------------
// 3) keys -> bf16 (row major) and bf16 transposed (D x M)
// ---------------------------------------------------------------------------
__global__ void k_keys_cvt(const float* __restrict__ keys, bf16_t* __restrict__ kb,
                           bf16_t* __restrict__ kt) {
    int idx = blockIdx.x * blockDim.x + threadIdx.x;      // over M_*D_ = 262144
    if (idx >= M_ * D_) return;
    int d = idx % D_;
    int m = idx / D_;
    float v = keys[idx];
    kb[idx]                = (bf16_t)v;
    kt[(size_t)d * M_ + m] = (bf16_t)v;
}

// ---------------------------------------------------------------------------
// 4) NT GEMM, bf16 WMMA, f32 accumulate:  C[i,j] = sum_k A[i,k] * BT[j,k]
//    A: n x 512 row-major bf16, BT: N x 512 row-major bf16, C: n x 512 f32.
//    Tile 128x128, BK=32, 256 threads (8 waves), each wave 2x4 WMMA tiles.
//    Tiles staged by the Tensor Data Mover, double-buffered in LDS.
// ---------------------------------------------------------------------------
#define BM 128
#define BN 128
#define BK 32
#define KTOT 512
#define KITERS (KTOT / BK)

__global__ __launch_bounds__(256)
void k_gemm_nt_bf16(const bf16_t* __restrict__ A, const bf16_t* __restrict__ BT,
                    float* __restrict__ Cp) {
    __shared__ __align__(16) bf16_t sA[2][BM * BK];
    __shared__ __align__(16) bf16_t sB[2][BN * BK];

    const int tid     = threadIdx.x;
    const int lane    = tid & 31;
    const int wave    = tid >> 5;
    const int waveRow = wave >> 1;   // 0..3 -> 32-row band
    const int waveCol = wave & 1;    // 0..1 -> 64-col band
    const int halfSel = lane >> 4;   // 0/1: lane half selects K sub-group
    const int l15     = lane & 15;

    const size_t rowBase = (size_t)blockIdx.y * BM;
    const size_t colBase = (size_t)blockIdx.x * BN;
    const bf16_t* aBase  = A  + rowBase * KTOT;
    const bf16_t* bBase  = BT + colBase * KTOT;

    v8f acc[2][4];
    #pragma unroll
    for (int r = 0; r < 2; ++r)
        #pragma unroll
        for (int c = 0; c < 4; ++c) acc[r][c] = v8f_zero();

    // prologue: DMA first tile pair into buffer 0
    if (wave == 0) {
        tdm_load_tile_128x32(aBase, lds_offset_of(&sA[0][0]));
        tdm_load_tile_128x32(bBase, lds_offset_of(&sB[0][0]));
        __builtin_amdgcn_s_wait_tensorcnt(0);
    }
    __syncthreads();

    for (int it = 0; it < KITERS; ++it) {
        const int cur = it & 1;

        // issue DMA for the next tile pair into the other buffer
        if (wave == 0 && it + 1 < KITERS) {
            const int k0n = (it + 1) * BK;
            tdm_load_tile_128x32(aBase + k0n, lds_offset_of(&sA[cur ^ 1][0]));
            tdm_load_tile_128x32(bBase + k0n, lds_offset_of(&sB[cur ^ 1][0]));
        }

        // A fragments: lane half 0 -> K {0..7,16..23}, half 1 -> K {8..15,24..31}
        FragBF aF[2], bF[4];
        #pragma unroll
        for (int r = 0; r < 2; ++r) {
            int ar = waveRow * 32 + r * 16 + l15;
            aF[r].u[0] = *reinterpret_cast<const uint4*>(&sA[cur][ar * BK + halfSel * 8]);
            aF[r].u[1] = *reinterpret_cast<const uint4*>(&sA[cur][ar * BK + 16 + halfSel * 8]);
        }
        // B fragments (transposed-operand tile): half 0 -> K {0..7,16..23}, half 1 -> K {8..15,24..31}
        #pragma unroll
        for (int c = 0; c < 4; ++c) {
            int bc = waveCol * 64 + c * 16 + l15;
            bF[c].u[0] = *reinterpret_cast<const uint4*>(&sB[cur][bc * BK + halfSel * 8]);
            bF[c].u[1] = *reinterpret_cast<const uint4*>(&sB[cur][bc * BK + 16 + halfSel * 8]);
        }

        #pragma unroll
        for (int r = 0; r < 2; ++r)
            #pragma unroll
            for (int c = 0; c < 4; ++c)
                acc[r][c] = __builtin_amdgcn_wmma_f32_16x16x32_bf16(
                    false, aF[r].v, false, bF[c].v, (short)0, acc[r][c], false, false);

        // make sure the next-buffer DMA landed before anyone proceeds
        if (wave == 0) __builtin_amdgcn_s_wait_tensorcnt(0);
        __syncthreads();
    }

    // D layout: lane -> col = l15, rows = halfSel*8 + v
    const int mOff = halfSel * 8;
    #pragma unroll
    for (int r = 0; r < 2; ++r)
        #pragma unroll
        for (int c = 0; c < 4; ++c) {
            size_t col = colBase + waveCol * 64 + c * 16 + l15;
            size_t rw  = rowBase + waveRow * 32 + r * 16 + mOff;
            #pragma unroll
            for (int v = 0; v < 8; ++v)
                Cp[(rw + v) * 512 + col] = acc[r][c][v];
        }
}

// ---------------------------------------------------------------------------
// 5) row softmax + top2 (wave per row). On entry sm_slot holds raw score.
//    Writes: raw copy -> sq_slot, softmax -> sm_slot, bf16 softmax -> sm_bf,
//    top-2 indices, and sraw[i] = score[i, top1] (row max).
// ---------------------------------------------------------------------------
__global__ __launch_bounds__(256)
void k_row_softmax(float* __restrict__ sm_slot, float* __restrict__ sq_slot,
                   bf16_t* __restrict__ sm_bf, int* __restrict__ gi0,
                   int* __restrict__ gi1, float* __restrict__ sraw) {
    const int lane = threadIdx.x & 31;
    const int wave = threadIdx.x >> 5;
    const int row  = blockIdx.x * 8 + wave;
    float* rp = sm_slot + (size_t)row * M_;

    float v[16];
    float v1 = -3.4e38f, v2 = -3.4e38f;
    int   i1 = 0, i2 = 0;
    #pragma unroll
    for (int j = 0; j < 16; ++j) {
        int m = lane + 32 * j;
        float x = rp[m];
        v[j] = x;
        if (x > v1)      { v2 = v1; i2 = i1; v1 = x; i1 = m; }
        else if (x > v2) { v2 = x; i2 = m; }
    }
    #pragma unroll
    for (int off = 16; off >= 1; off >>= 1) {
        float w1 = __shfl_xor(v1, off);
        int   j1 = __shfl_xor(i1, off);
        float w2 = __shfl_xor(v2, off);
        int   j2 = __shfl_xor(i2, off);
        if (w1 > v1) {
            float c = v1; int ci = i1;
            v1 = w1; i1 = j1;
            if (w2 > c) { v2 = w2; i2 = j2; } else { v2 = c; i2 = ci; }
        } else if (w1 > v2) { v2 = w1; i2 = j1; }
    }
    const float mx = v1;
    float s = 0.f;
    #pragma unroll
    for (int j = 0; j < 16; ++j) s += expf(v[j] - mx);
    #pragma unroll
    for (int off = 16; off >= 1; off >>= 1) s += __shfl_xor(s, off);
    const float inv = 1.0f / s;

    float*  qp = sq_slot + (size_t)row * M_;
    bf16_t* bp = sm_bf   + (size_t)row * M_;
    #pragma unroll
    for (int j = 0; j < 16; ++j) {
        int m = lane + 32 * j;
        float e = expf(v[j] - mx) * inv;
        qp[m] = v[j];          // raw copy for the column softmax
        rp[m] = e;             // score_memory
        bp[m] = (bf16_t)e;     // bf16 operand for GEMM2
    }
    if (lane == 0) { gi0[row] = i1; gi1[row] = i2; sraw[row] = v1; }
}

// ---------------------------------------------------------------------------
// 6) column softmax helpers (thread <-> column, coalesced)
// ---------------------------------------------------------------------------
__global__ __launch_bounds__(512)
void k_col_max_part(const float* __restrict__ raw, float* __restrict__ pmax) {
    const int col = threadIdx.x;
    const int r0  = blockIdx.x * 256;
    float m = -3.4e38f;
    for (int r = 0; r < 256; ++r)
        m = fmaxf(m, raw[(size_t)(r0 + r) * M_ + col]);
    pmax[(size_t)blockIdx.x * M_ + col] = m;
}

__global__ __launch_bounds__(512)
void k_col_max_fin(const float* __restrict__ pmax, float* __restrict__ colmax) {
    const int col = threadIdx.x;
    float m = -3.4e38f;
    for (int c = 0; c < 128; ++c) m = fmaxf(m, pmax[(size_t)c * M_ + col]);
    colmax[col] = m;
}

__global__ __launch_bounds__(512)
void k_col_sum_part(const float* __restrict__ raw, const float* __restrict__ colmax,
                    float* __restrict__ psum) {
    const int col = threadIdx.x;
    const int r0  = blockIdx.x * 256;
    const float cm = colmax[col];
    float s = 0.f;
    for (int r = 0; r < 256; ++r)
        s += expf(raw[(size_t)(r0 + r) * M_ + col] - cm);
    psum[(size_t)blockIdx.x * M_ + col] = s;
}

__global__ __launch_bounds__(512)
void k_col_sum_fin(const float* __restrict__ psum, float* __restrict__ colsum) {
    const int col = threadIdx.x;
    float s = 0.f;
    for (int c = 0; c < 128; ++c) s += psum[(size_t)c * M_ + col];
    colsum[col] = s;
}

__global__ void k_sq_fin(float* __restrict__ sq, const float* __restrict__ colmax,
                         const float* __restrict__ colsum) {
    int idx = blockIdx.x * blockDim.x + threadIdx.x;   // 16777216
    if (idx >= NR_ * M_) return;
    int m = idx & (M_ - 1);
    sq[idx] = expf(sq[idx] - colmax[m]) / colsum[m];
}

// ---------------------------------------------------------------------------
// 7) losses + weighted segment sum (block per row)
// ---------------------------------------------------------------------------
__global__ __launch_bounds__(256)
void k_loss_seg(const float* __restrict__ q, const float* __restrict__ invn,
                const float* __restrict__ keys, const int* __restrict__ gi0,
                const int* __restrict__ gi1, const float* __restrict__ sraw,
                const float* __restrict__ colmax, float* __restrict__ qupd,
                float* __restrict__ acc) {
    const int i   = blockIdx.x;
    const int tid = threadIdx.x;
    const int g0  = gi0[i], g1 = gi1[i];
    const int t   = i % T_;
    const int b   = i / (C_ * T_);
    const float* qrow  = q    + (size_t)i * D_;
    const float* inrow = invn + ((size_t)(b * T_ + t)) * D_;

    float qr[2];
    float comp = 0.f, dp = 0.f, dn = 0.f;
    #pragma unroll
    for (int e = 0; e < 2; ++e) {
        int d = tid + e * 256;
        float v  = qrow[d] * inrow[d];
        qr[e] = v;
        float p  = keys[(size_t)g0 * D_ + d];
        float ng = keys[(size_t)g1 * D_ + d];
        float dc = v - p;           comp += dc * dc;
        float a  = dc + 1e-6f;      dp   += a * a;
        float bb = v - ng + 1e-6f;  dn   += bb * bb;
    }
    __shared__ float s0[256], s1[256], s2[256];
    s0[tid] = comp; s1[tid] = dp; s2[tid] = dn;
    __syncthreads();
    for (int off = 128; off > 0; off >>= 1) {
        if (tid < off) {
            s0[tid] += s0[tid + off];
            s1[tid] += s1[tid + off];
            s2[tid] += s2[tid + off];
        }
        __syncthreads();
    }
    if (tid == 0) {
        atomicAdd(&acc[0], s0[0]);                                  // compactness sum
        float sep = fmaxf(sqrtf(s1[0]) - sqrtf(s2[0]) + 1.0f, 0.f); // margin term
        atomicAdd(&acc[1], sep);
    }
    // w_i = score_query[i,top1]/colmax(score_query)[top1] = exp(raw_top1 - colmax_raw)
    const float w = expf(sraw[i] - colmax[g0]);
    #pragma unroll
    for (int e = 0; e < 2; ++e)
        atomicAdd(&qupd[(size_t)g0 * D_ + tid + e * 256], w * qr[e]);
}

// ---------------------------------------------------------------------------
// 8) updated_memory = l2norm(query_update + keys) per row
// ---------------------------------------------------------------------------
__global__ __launch_bounds__(256)
void k_upd_mem(const float* __restrict__ qupd, const float* __restrict__ keys,
               float* __restrict__ outm) {
    const int m = blockIdx.x, tid = threadIdx.x;
    float a0 = qupd[(size_t)m * D_ + tid]       + keys[(size_t)m * D_ + tid];
    float a1 = qupd[(size_t)m * D_ + tid + 256] + keys[(size_t)m * D_ + tid + 256];
    __shared__ float s[256];
    s[tid] = a0 * a0 + a1 * a1;
    __syncthreads();
    for (int off = 128; off > 0; off >>= 1) {
        if (tid < off) s[tid] += s[tid + off];
        __syncthreads();
    }
    __shared__ float rn;
    if (tid == 0) rn = 1.0f / fmaxf(sqrtf(s[0]), 1e-12f);
    __syncthreads();
    outm[(size_t)m * D_ + tid]       = a0 * rn;
    outm[(size_t)m * D_ + tid + 256] = a1 * rn;
}

__global__ void k_fin_loss(const float* __restrict__ acc, float* __restrict__ out_sep,
                           float* __restrict__ out_comp) {
    if (threadIdx.x == 0 && blockIdx.x == 0) {
        out_sep[0]  = acc[1] / (float)NR_;
        out_comp[0] = acc[0] / ((float)NR_ * (float)D_);
    }
}

// ---------------------------------------------------------------------------
extern "C" void kernel_launch(void* const* d_in, const int* in_sizes, int n_in,
                              void* d_out, int out_size, void* d_ws, size_t ws_size,
                              hipStream_t stream) {
    const float* query = (const float*)d_in[0];   // 16*64*32*512
    const float* keys  = (const float*)d_in[1];   // 512*512

    float* out_updq = (float*)d_out;              // 16777216
    float* out_mem  = out_updq + 16777216;        // 262144
    float* out_sq   = out_mem  + 262144;          // 16777216
    float* out_sm   = out_sq   + 16777216;        // 16777216
    float* out_sep  = out_sm   + 16777216;        // 1
    float* out_comp = out_sep  + 1;               // 1

    char* w = (char*)d_ws;                        // ~71.2 MB used
    bf16_t* qr_bf  = (bf16_t*)(w + 0);            // 32 MB
    bf16_t* sm_bf  = (bf16_t*)(w + 33554432);     // 32 MB
    bf16_t* kb     = (bf16_t*)(w + 67108864);     // 512 KB
    bf16_t* kt     = (bf16_t*)(w + 67633152);     // 512 KB
    float*  invn   = (float*) (w + 68157440);     // 1 MB
    float*  pmax   = (float*) (w + 69206016);     // 256 KB
    float*  psum   = (float*) (w + 69468160);     // 256 KB
    float*  colmax = (float*) (w + 69730304);     // 2 KB
    float*  colsum = (float*) (w + 69732352);     // 2 KB
    int*    gi0    = (int*)   (w + 69734400);     // 128 KB
    int*    gi1    = (int*)   (w + 69865472);     // 128 KB
    float*  sraw   = (float*) (w + 69996544);     // 128 KB
    float*  qupd   = (float*) (w + 70127616);     // 1 MB (+8B acc follows)
    float*  acc    = (float*) (w + 71176192);     // 2 floats

    // zero query_update + accumulators (contiguous 262146 floats)
    k_zero<<<1024, 256, 0, stream>>>(qupd, 262146);

    k_inv_norm<<<(B_*T_*D_ + 255) / 256, 256, 0, stream>>>(query, invn);
    k_qr_bf16 <<<(NR_*D_) / 256, 256, 0, stream>>>(query, invn, qr_bf);
    k_keys_cvt<<<(M_*D_) / 256, 256, 0, stream>>>(keys, kb, kt);

    // GEMM1: raw score -> score_memory slot
    dim3 gGemm(M_ / BN, NR_ / BM);                // (4, 256)
    k_gemm_nt_bf16<<<gGemm, 256, 0, stream>>>(qr_bf, kb, out_sm);

    // row softmax + top2; raw copy into score_query slot
    k_row_softmax<<<NR_ / 8, 256, 0, stream>>>(out_sm, out_sq, sm_bf, gi0, gi1, sraw);

    // GEMM2: updated_query = score_memory @ keys
    k_gemm_nt_bf16<<<gGemm, 256, 0, stream>>>(sm_bf, kt, out_updq);

    // column softmax over raw score in score_query slot
    k_col_max_part<<<NR_ / 256, 512, 0, stream>>>(out_sq, pmax);
    k_col_max_fin <<<1, 512, 0, stream>>>(pmax, colmax);
    k_col_sum_part<<<NR_ / 256, 512, 0, stream>>>(out_sq, colmax, psum);
    k_col_sum_fin <<<1, 512, 0, stream>>>(psum, colsum);

    // losses + segment sum (needs colmax, raw top1)
    k_loss_seg<<<NR_, 256, 0, stream>>>(query, invn, keys, gi0, gi1, sraw,
                                        colmax, qupd, acc);

    // finalize score_query in place
    k_sq_fin<<<(NR_*M_) / 256, 256, 0, stream>>>(out_sq, colmax, colsum);

    k_upd_mem<<<M_, 256, 0, stream>>>(qupd, keys, out_mem);
    k_fin_loss<<<1, 1, 0, stream>>>(acc, out_sep, out_comp);
}
